// Attention_36369783062801
// MI455X (gfx1250) — matrix operand order repaired and checked by reading, not compiled, not run on hardware
//
#include <hip/hip_runtime.h>

// ---------------------------------------------------------------------------
// CDNA5 (gfx1250) fused attention:  cvt -> QKV gemm -> flash-attn -> out gemm
// All matmuls via v_wmma_f32_16x16x32_f16 (wave32, fp32 accumulate).
// fp32 operands pre-converted to f16 once; K is produced transposed [b,h,d,s]
// and attention fragments are gathered per element (K strided over d).
// ---------------------------------------------------------------------------

typedef __attribute__((ext_vector_type(16))) _Float16 v16h;
typedef __attribute__((ext_vector_type(8)))  _Float16 v8h;
typedef __attribute__((ext_vector_type(8)))  float    v8f;

#define BATCH    2
#define SEQ      2048
#define DIM      1024
#define HEADS    16
#define DIM_HEAD 64
#define INNER    1024   // HEADS * DIM_HEAD

// A-fragment K offset for 16-bit WMMA A (16x32):
// lanes 0-15 hold K {0..7,16..23}, lanes 16-31 hold K {8..15,24..31}
__device__ __forceinline__ int a_kofs(int e, int half) {
  return ((e < 8) ? e : (e + 8)) + (half << 3);
}

__device__ __forceinline__ v8f wmma_f16(v16h a, v16h b, v8f c) {
  return __builtin_amdgcn_wmma_f32_16x16x32_f16(
      /*neg_a=*/false, a, /*neg_b=*/false, b,
      /*c_mod=*/(short)0, c, /*reuse_a=*/false, /*reuse_b=*/false);
}

// Build a 16x32 f16 A-fragment from two contiguous 8-half chunks.
__device__ __forceinline__ v16h load_a_f16(const _Float16* Arow, int k0, int half) {
  v8h lo = *(const v8h*)(Arow + k0 + half * 8);
  v8h hi = *(const v8h*)(Arow + k0 + 16 + half * 8);
  v16h a;
#pragma unroll
  for (int i = 0; i < 8; ++i) { a[i] = lo[i]; a[i + 8] = hi[i]; }
  return a;
}

// ---------------------------------------------------------------------------
// Kernel 0: fp32 -> f16 elementwise convert (8 elems/thread, exact grids).
// ---------------------------------------------------------------------------
__global__ __launch_bounds__(256) void cvt_f16_kernel(
    const float* __restrict__ in, _Float16* __restrict__ out)
{
  const size_t i = ((size_t)blockIdx.x * 256 + threadIdx.x) * 8;
  float4 f0 = *(const float4*)(in + i);
  float4 f1 = *(const float4*)(in + i + 4);
  v8h h;
  h[0] = (_Float16)f0.x; h[1] = (_Float16)f0.y;
  h[2] = (_Float16)f0.z; h[3] = (_Float16)f0.w;
  h[4] = (_Float16)f1.x; h[5] = (_Float16)f1.y;
  h[6] = (_Float16)f1.z; h[7] = (_Float16)f1.w;
  *(v8h*)(out + i) = h;
}

// ---------------------------------------------------------------------------
// Kernel 1: QKV projection.  which=0: Q = Xq @ Wq ; which=1/2: K/V = Xkv @ Wk/v
// One wave per 32x64 C tile; all-f16 loads.
// Q,V written [b,h,s,d]; K written transposed [b,h,d,s] for the attn kernel.
// 3 * 128 * 16 = 6144 wave-tasks, 8 waves/block -> 768 blocks.
// ---------------------------------------------------------------------------
__global__ __launch_bounds__(256) void qkv_proj_kernel(
    const _Float16* __restrict__ Xq, const _Float16* __restrict__ Xkv,
    const _Float16* __restrict__ Wh,   // [K=1024, N=3072] f16
    _Float16* __restrict__ Qb, _Float16* __restrict__ Kt,
    _Float16* __restrict__ Vb)
{
  const int lane  = threadIdx.x & 31;
  const int wv    = threadIdx.x >> 5;
  const int task  = blockIdx.x * 8 + wv;     // 0..6143
  const int which = task >> 11;              // 0..2
  const int t2    = task & 2047;
  const int m0    = (t2 >> 4) << 5;          // 32-row base (0..4064)
  const int n0    = (t2 & 15) << 6;          // 64-col base (0..960)
  const int half  = lane >> 4;
  const int r     = lane & 15;

  const _Float16* X    = (which == 0) ? Xq : Xkv;
  const _Float16* A0   = X + (size_t)(m0 + r) * DIM;   // [b,s,dim] flat == row m
  const _Float16* A1   = A0 + (size_t)16 * DIM;
  const _Float16* Bcol = Wh + (size_t)which * INNER + n0;

  v8f c[2][4] = {};
  for (int k0 = 0; k0 < DIM; k0 += 32) {
    v16h a[2];
    a[0] = load_a_f16(A0, k0, half);
    a[1] = load_a_f16(A1, k0, half);
    const _Float16* Brow = Bcol + r;
#pragma unroll
    for (int nt = 0; nt < 4; ++nt) {
      v16h b;
#pragma unroll
      for (int e = 0; e < 16; ++e)
        b[e] = Brow[(size_t)(k0 + a_kofs(e, half)) * (3 * INNER) + nt * 16];
      c[0][nt] = wmma_f16(a[0], b, c[0][nt]);
      c[1][nt] = wmma_f16(a[1], b, c[1][nt]);
    }
  }

  _Float16* dst = (which == 0) ? Qb : (which == 1) ? Kt : Vb;
#pragma unroll
  for (int mt = 0; mt < 2; ++mt)
#pragma unroll
    for (int nt = 0; nt < 4; ++nt)
#pragma unroll
      for (int e = 0; e < 8; ++e) {
        const int m = m0 + mt * 16 + e + half * 8;   // global row in [0,4096)
        const int n = n0 + nt * 16 + r;              // inner index in [0,1024)
        const int batch = m >> 11, s = m & 2047;
        const int h = n >> 6, d = n & 63;
        const size_t bh = (size_t)batch * HEADS + h;
        const size_t idx = (which == 1)
            ? (bh * DIM_HEAD + d) * SEQ + s          // K: [b,h,d,s]
            : ((bh * SEQ + s) << 6) + d;             // Q,V: [b,h,s,d]
        dst[idx] = (_Float16)c[mt][nt][e];
      }
}

// ---------------------------------------------------------------------------
// Kernel 2: FlashAttention.  One wave per (b,h, 32-row q-tile); kv step 32.
// K/V fragments are strided per-element gathers, shared across the two M tiles.
// 2*16*64 = 2048 wave-tasks, 4 waves/block -> 512 blocks.
// ---------------------------------------------------------------------------
__global__ __launch_bounds__(128) void attn_kernel(
    const _Float16* __restrict__ Qb, const _Float16* __restrict__ Ktb,
    const _Float16* __restrict__ Vb, _Float16* __restrict__ Ob)
{
  __shared__ _Float16 smem[4][2][16 * 32];  // per-wave, per-Mtile 16x32 P tile
  const int lane = threadIdx.x & 31;
  const int wv   = threadIdx.x >> 5;
  const int task = blockIdx.x * 4 + wv;     // 0..2047
  const int qt   = task & 63;
  const int bh   = task >> 6;               // 0..31
  const int b    = bh >> 4;
  const int h    = bh & 15;
  const int q0   = qt << 5;                 // 32-row q base
  const int half = lane >> 4;
  const int r    = lane & 15;

  const size_t base = (size_t)bh * SEQ * DIM_HEAD;
  const _Float16* Q  = Qb  + base;
  const _Float16* Kt = Ktb + base;          // [d, s] per (b,h)
  const _Float16* V  = Vb  + base;          // [s, d] per (b,h)

  const float scale = 0.125f;  // DIM_HEAD^-0.5

  // Q fragments: [mtile][kchunk], rows q0+mt*16+r, pre-scaled.
  v16h qa[2][2];
#pragma unroll
  for (int mt = 0; mt < 2; ++mt)
#pragma unroll
    for (int kc = 0; kc < 2; ++kc)
#pragma unroll
      for (int e = 0; e < 16; ++e)
        qa[mt][kc][e] = (_Float16)(
            (float)Q[(size_t)(q0 + mt * 16 + r) * 64 + kc * 32 +
                     a_kofs(e, half)] * scale);

  v8f o[2][4] = {};
  float rmax[2][8], rsum[2][8];
#pragma unroll
  for (int mt = 0; mt < 2; ++mt)
#pragma unroll
    for (int e = 0; e < 8; ++e) { rmax[mt][e] = -1e30f; rsum[mt][e] = 0.f; }

  for (int j = 0; j < SEQ; j += 32) {
    // K fragments [stile][kchunk]: B[kd, n] = Kt[d = kc*32+a_kofs(e,half)][kv = j+st*16+r]
    // strided gather over d.
    const _Float16* krow = Kt + j + r;
    v16h kb[2][2];
#pragma unroll
    for (int st = 0; st < 2; ++st)
#pragma unroll
      for (int kc = 0; kc < 2; ++kc)
#pragma unroll
        for (int e = 0; e < 16; ++e)
          kb[st][kc][e] = krow[(size_t)(kc * 32 + a_kofs(e, half)) * SEQ + st * 16];

    // S = Q @ K^T : 2 Mtiles x 2 Stiles
    v8f s[2][2] = {};
#pragma unroll
    for (int kc = 0; kc < 2; ++kc)
#pragma unroll
      for (int mt = 0; mt < 2; ++mt)
#pragma unroll
        for (int st = 0; st < 2; ++st)
          s[mt][st] = wmma_f16(qa[mt][kc], kb[st][kc], s[mt][st]);

    // Online softmax (rows = C element e + 8*half; 16 cols across half-lanes)
#pragma unroll
    for (int mt = 0; mt < 2; ++mt) {
      _Float16* pm = smem[wv][mt];
#pragma unroll
      for (int e = 0; e < 8; ++e) {
        float v0 = s[mt][0][e], v1 = s[mt][1][e];
        float mx = fmaxf(v0, v1);
#pragma unroll
        for (int msk = 8; msk >= 1; msk >>= 1)
          mx = fmaxf(mx, __shfl_xor(mx, msk, 32));
        const float mnew = fmaxf(rmax[mt][e], mx);
        const float corr = __expf(rmax[mt][e] - mnew);
        rmax[mt][e] = mnew;
        const float p0 = __expf(v0 - mnew);
        const float p1 = __expf(v1 - mnew);
        float rs = p0 + p1;
#pragma unroll
        for (int msk = 8; msk >= 1; msk >>= 1)
          rs += __shfl_xor(rs, msk, 32);
        rsum[mt][e] = rsum[mt][e] * corr + rs;
        o[mt][0][e] *= corr; o[mt][1][e] *= corr;
        o[mt][2][e] *= corr; o[mt][3][e] *= corr;
        const int row = e + half * 8;
        pm[row * 32 + r]      = (_Float16)p0;
        pm[row * 32 + 16 + r] = (_Float16)p1;
      }
    }
    asm volatile("s_wait_dscnt 0" ::: "memory");  // C-layout -> A-layout via LDS

    v16h pa[2];
#pragma unroll
    for (int mt = 0; mt < 2; ++mt)
#pragma unroll
      for (int e = 0; e < 16; ++e)
        pa[mt][e] = smem[wv][mt][r * 32 + a_kofs(e, half)];

    // O += P @ V : shared V B-frags (lane = col r, elements = kv rows j+a_kofs(e,half))
    const _Float16* vrow = V + (size_t)j * 64 + r;
    v16h vb0, vb1, vb2, vb3;
#pragma unroll
    for (int e = 0; e < 16; ++e) {
      const size_t vo = (size_t)a_kofs(e, half) * 64;
      vb0[e] = vrow[vo + 0]; vb1[e] = vrow[vo + 16];
      vb2[e] = vrow[vo + 32]; vb3[e] = vrow[vo + 48];
    }
#pragma unroll
    for (int mt = 0; mt < 2; ++mt) {
      o[mt][0] = wmma_f16(pa[mt], vb0, o[mt][0]);
      o[mt][1] = wmma_f16(pa[mt], vb1, o[mt][1]);
      o[mt][2] = wmma_f16(pa[mt], vb2, o[mt][2]);
      o[mt][3] = wmma_f16(pa[mt], vb3, o[mt][3]);
    }
  }

  // Normalize and store f16 [b, s, h*64+d]
#pragma unroll
  for (int mt = 0; mt < 2; ++mt)
#pragma unroll
    for (int e = 0; e < 8; ++e) {
      const float inv = 1.0f / rsum[mt][e];
      const int s = q0 + mt * 16 + e + half * 8;
      const size_t orow = ((size_t)b * SEQ + s) * INNER + h * DIM_HEAD;
      Ob[orow + 0  + r] = (_Float16)(o[mt][0][e] * inv);
      Ob[orow + 16 + r] = (_Float16)(o[mt][1][e] * inv);
      Ob[orow + 32 + r] = (_Float16)(o[mt][2][e] * inv);
      Ob[orow + 48 + r] = (_Float16)(o[mt][3][e] * inv);
    }
}

// ---------------------------------------------------------------------------
// Kernel 3: output projection  out = Ob(f16) @ Wout(f16) + b_out  (fp32 store)
// One wave per 32x64 C tile. 128*16 = 2048 wave-tasks, 8 waves/block -> 256.
// ---------------------------------------------------------------------------
__global__ __launch_bounds__(256) void out_proj_kernel(
    const _Float16* __restrict__ Ob, const _Float16* __restrict__ Wh,
    const float* __restrict__ b_out, float* __restrict__ out)
{
  const int lane = threadIdx.x & 31;
  const int wv   = threadIdx.x >> 5;
  const int task = blockIdx.x * 8 + wv;     // 0..2047
  const int m0   = (task >> 4) << 5;
  const int n0   = (task & 15) << 6;
  const int half = lane >> 4;
  const int r    = lane & 15;

  const _Float16* A0 = Ob + (size_t)(m0 + r) * INNER;
  const _Float16* A1 = A0 + (size_t)16 * INNER;

  v8f c[2][4] = {};
  for (int k0 = 0; k0 < INNER; k0 += 32) {
    v16h a[2];
    a[0] = load_a_f16(A0, k0, half);
    a[1] = load_a_f16(A1, k0, half);
    const _Float16* Brow = Wh + n0 + r;
#pragma unroll
    for (int nt = 0; nt < 4; ++nt) {
      v16h b;
#pragma unroll
      for (int e = 0; e < 16; ++e)
        b[e] = Brow[(size_t)(k0 + a_kofs(e, half)) * DIM + nt * 16];
      c[0][nt] = wmma_f16(a[0], b, c[0][nt]);
      c[1][nt] = wmma_f16(a[1], b, c[1][nt]);
    }
  }

#pragma unroll
  for (int mt = 0; mt < 2; ++mt)
#pragma unroll
    for (int nt = 0; nt < 4; ++nt)
#pragma unroll
      for (int e = 0; e < 8; ++e) {
        const int m = m0 + mt * 16 + e + half * 8;
        const int n = n0 + nt * 16 + r;
        out[(size_t)m * DIM + n] = c[mt][nt][e] + b_out[n];
      }
}

// ---------------------------------------------------------------------------
extern "C" void kernel_launch(void* const* d_in, const int* in_sizes, int n_in,
                              void* d_out, int out_size, void* d_ws, size_t ws_size,
                              hipStream_t stream) {
  const float* x_q   = (const float*)d_in[0];
  const float* x_kv  = (const float*)d_in[1];
  const float* W_qkv = (const float*)d_in[2];
  const float* W_out = (const float*)d_in[3];
  const float* b_out = (const float*)d_in[4];
  float* out = (float*)d_out;

  // f16 workspace layout (element offsets; 4 Mi = 4194304)
  const size_t M4 = (size_t)4 * 1024 * 1024;
  _Float16* Qb   = (_Float16*)d_ws;        // [0,  4Mi)
  _Float16* Ktb  = Qb + M4;                // [4,  8Mi)  K transposed [b,h,d,s]
  _Float16* Vb   = Ktb + M4;               // [8, 12Mi)
  _Float16* Xqh  = Vb + M4;                // [12,16Mi)  (aliased as Ob later)
  _Float16* Xkvh = Xqh + M4;               // [16,20Mi)
  _Float16* Wqkvh= Xkvh + M4;              // [20,23Mi)  3 Mi
  _Float16* Wouth= Wqkvh + (size_t)3 * 1024 * 1024; // [23,24Mi) 1 Mi
  _Float16* Ob   = Xqh;                    // dead after kernel 1 -> reuse

  // Pre-convert all fp32 operands to f16 (exact grids, 2048 elems/block).
  cvt_f16_kernel<<<2048, 256, 0, stream>>>(x_q,   Xqh);
  cvt_f16_kernel<<<2048, 256, 0, stream>>>(x_kv,  Xkvh);
  cvt_f16_kernel<<<1536, 256, 0, stream>>>(W_qkv, Wqkvh);
  cvt_f16_kernel<<< 512, 256, 0, stream>>>(W_out, Wouth);

  qkv_proj_kernel<<<768, 256, 0, stream>>>(Xqh, Xkvh, Wqkvh, Qb, Ktb, Vb);
  attn_kernel<<<512, 128, 0, stream>>>(Qb, Ktb, Vb, Ob);
  out_proj_kernel<<<256, 256, 0, stream>>>(Ob, Wouth, b_out, out);
}
